// FeatureInteractionProcess_80659485819393
// MI455X (gfx1250) — matrix operand order repaired and checked by reading, not compiled, not run on hardware
//
#include <hip/hip_runtime.h>
#include <hip/hip_bf16.h>

// ---------------- problem constants ----------------
#define BB   4
#define CC   64
#define HH   64           // high res
#define HL   128          // low res
#define HW   (HL*HL)      // 16384 pixels per plane
#define NT   (HW/16)      // 1024 N-tiles (16 pixels each) per batch
#define KK3  27           // 3*3*3 offset-mask channels

typedef __bf16  bf16_t;
typedef __bf16    v16bf __attribute__((ext_vector_type(16)));
typedef float     v8f   __attribute__((ext_vector_type(8)));
typedef unsigned  v8u   __attribute__((ext_vector_type(8)));

// scalar f32 -> bf16 (RNE) for the cold pack kernel
static __device__ __forceinline__ bf16_t f2bf(float f) {
    unsigned u = __float_as_uint(f);
    unsigned r = u + 0x7FFFu + ((u >> 16) & 1u);
    unsigned short h = (unsigned short)(r >> 16);
    return __builtin_bit_cast(bf16_t, h);
}

// packed pair f32x2 -> bf16x2 in one dword (v_cvt_pk_bf16_f32 when available)
static __device__ __forceinline__ unsigned pack2bf(float lo, float hi) {
#if __has_builtin(__builtin_amdgcn_cvt_pk_bf16_f32)
    typedef __bf16 v2bf __attribute__((ext_vector_type(2)));
    v2bf p = __builtin_amdgcn_cvt_pk_bf16_f32(lo, hi);
    return __builtin_bit_cast(unsigned, p);
#else
    unsigned a = (unsigned)__builtin_bit_cast(unsigned short, (bf16_t)lo);
    unsigned b = (unsigned)__builtin_bit_cast(unsigned short, (bf16_t)hi);
    return a | (b << 16);
#endif
}

static __device__ __forceinline__ v8f wmma_bf16(v16bf a, v16bf b, v8f c) {
    // D(f32 16x16) = A(bf16 16x32) * B(bf16 32x16) + C
    return __builtin_amdgcn_wmma_f32_16x16x32_bf16(false, a, false, b,
                                                   (short)0, c, false, false);
}

// ---------------------------------------------------------------------------
// K0: pack weight matrix (realM x Kdim, row-major) into WMMA A-fragment order:
// frag[(mt*Ktiles+kt)*32 + lane][e]; this file's contiguous k order (B matches):
//   h = lane>>4, m = 16*mt + (lane&15)
//   k = 32*kt + 16*h + e
// ---------------------------------------------------------------------------
__global__ void k_pack(const float* __restrict__ src, bf16_t* __restrict__ dst,
                       int Mtiles, int Ktiles, int realM, int Kdim) {
    int el = blockIdx.x * blockDim.x + threadIdx.x;
    int total = Mtiles * Ktiles * 512;
    if (el >= total) return;
    int tile = el >> 9, w = el & 511;
    int lane = w >> 4, e = w & 15;
    int mt = tile / Ktiles, kt = tile % Ktiles;
    int h  = lane >> 4;
    int m  = mt * 16 + (lane & 15);
    int k  = kt * 32 + 16 * h + e;
    float v = (m < realM) ? src[m * Kdim + k] : 0.0f;
    dst[el] = f2bf(v);
}

// ---------------------------------------------------------------------------
// K1: align-corners bilinear upsample 64x64 -> 128x128 (per B,C plane)
// ---------------------------------------------------------------------------
__global__ void k_upsample(const float* __restrict__ high, float* __restrict__ up) {
    int idx = blockIdx.x * blockDim.x + threadIdx.x;      // B*C*HW threads
    int p  = idx & (HW - 1);
    int bc = idx >> 14;
    int y = p >> 7, x = p & (HL - 1);
    const float s = (float)(HH - 1) / (float)(HL - 1);    // 63/127
    float py = y * s, px = x * s;
    float y0f = floorf(py), x0f = floorf(px);
    int   y0 = (int)y0f,   x0 = (int)x0f;
    float fy = py - y0f,   fx = px - x0f;
    int   y1 = min(y0 + 1, HH - 1), x1 = min(x0 + 1, HH - 1);
    const float* sp = high + bc * (HH * HH);
    float v00 = sp[y0 * HH + x0], v01 = sp[y0 * HH + x1];
    float v10 = sp[y1 * HH + x0], v11 = sp[y1 * HH + x1];
    up[idx] = v00 * (1.f - fy) * (1.f - fx) + v01 * (1.f - fy) * fx
            + v10 * fy * (1.f - fx)         + v11 * fy * fx;
}

// ---------------------------------------------------------------------------
// K2: 1x1 conv as WMMA GEMM: ofeat[64,p] = Woff(64x128) . concat(hup,low)[128,p]
// one wave per (batch, 16-pixel tile); K = 128 = 4 WMMA steps; 4 M tiles
// ---------------------------------------------------------------------------
__global__ __launch_bounds__(256)
void k_conv1x1(const float* __restrict__ hup, const float* __restrict__ low,
               const bf16_t* __restrict__ afrag, float* __restrict__ ofeat) {
    int wave = blockIdx.x * 8 + (threadIdx.x >> 5);
    int lane = threadIdx.x & 31;
    int b = wave >> 10, nt = wave & (NT - 1);
    int nbase = nt * 16;
    int h = lane >> 4, n = lane & 15;
    int p = nbase + n;
    const v16bf* A = (const v16bf*)afrag;
    v8f acc[4] = {{}, {}, {}, {}};
    for (int kt = 0; kt < 4; ++kt) {
        v8u bw;
        #pragma unroll
        for (int j = 0; j < 8; ++j) {            // B layout: k = 32*kt + 16*h + e
            int k0 = kt * 32 + 16 * h + 2 * j;
            float v0 = (k0     < CC) ? hup[(b * CC + k0) * HW + p]
                                     : low[(b * CC + (k0 - CC)) * HW + p];
            float v1 = (k0 + 1 < CC) ? hup[(b * CC + k0 + 1) * HW + p]
                                     : low[(b * CC + (k0 + 1 - CC)) * HW + p];
            bw[j] = pack2bf(v0, v1);
        }
        v16bf bfr = __builtin_bit_cast(v16bf, bw);
        #pragma unroll
        for (int mt = 0; mt < 4; ++mt)
            acc[mt] = wmma_bf16(A[(mt * 4 + kt) * 32 + lane], bfr, acc[mt]);
    }
    #pragma unroll
    for (int mt = 0; mt < 4; ++mt)
        #pragma unroll
        for (int r = 0; r < 8; ++r) {            // D layout: m = 16*mt + 8*h + r
            int m = mt * 16 + 8 * h + r;
            ofeat[(b * CC + m) * HW + p] = acc[mt][r];
        }
}

// ---------------------------------------------------------------------------
// K3: 3x3 conv (pad 1) as WMMA GEMM: om[27,p] = Wom(27x576) . im2col(ofeat)
// K = 576 = 18 steps; M padded to 32 (2 tiles, rows >= 27 dropped on store)
// ---------------------------------------------------------------------------
__global__ __launch_bounds__(256)
void k_om(const float* __restrict__ ofeat, const bf16_t* __restrict__ afrag,
          const float* __restrict__ bias, float* __restrict__ om) {
    int wave = blockIdx.x * 8 + (threadIdx.x >> 5);
    int lane = threadIdx.x & 31;
    int b = wave >> 10, nt = wave & (NT - 1);
    int nbase = nt * 16;
    int h = lane >> 4, n = lane & 15;
    int p = nbase + n, y = p >> 7, x = p & (HL - 1);
    const float* fb = ofeat + b * CC * HW;
    const v16bf* A = (const v16bf*)afrag;
    v8f acc[2] = {{}, {}};
    for (int kt = 0; kt < 18; ++kt) {
        v8u bw;
        #pragma unroll
        for (int j = 0; j < 8; ++j) {
            float v[2];
            #pragma unroll
            for (int q = 0; q < 2; ++q) {
                int k = kt * 32 + 16 * h + 2 * j + q;
                int c = k / 9, t = k - 9 * c;
                int yy = y + t / 3 - 1, xx = x + t % 3 - 1;
                v[q] = (yy >= 0 && yy < HL && xx >= 0 && xx < HL)
                         ? fb[c * HW + yy * HL + xx] : 0.0f;
            }
            bw[j] = pack2bf(v[0], v[1]);
        }
        v16bf bfr = __builtin_bit_cast(v16bf, bw);
        acc[0] = wmma_bf16(A[(0 * 18 + kt) * 32 + lane], bfr, acc[0]);
        acc[1] = wmma_bf16(A[(1 * 18 + kt) * 32 + lane], bfr, acc[1]);
    }
    #pragma unroll
    for (int mt = 0; mt < 2; ++mt)
        #pragma unroll
        for (int r = 0; r < 8; ++r) {
            int m = mt * 16 + 8 * h + r;
            if (m < KK3) om[(b * KK3 + m) * HW + p] = acc[mt][r] + bias[m];
        }
}

// ---------------------------------------------------------------------------
// K4: deformable conv. Per wave: 16 pixels, all 64 out channels.
// Phase A: 144 tap records (9 taps x 16 pixels) -> LDS:
//          4 clamped corner plane-offsets + 4 validity*sigmoid(mask) weights.
// Phase B: K=576 loop; B-fragment = fused bilinear gather of x (bf16);
//          4 M tiles of v_wmma_f32_16x16x32_bf16; bias on store.
// ---------------------------------------------------------------------------
__global__ __launch_bounds__(256)
void k_dcn(const float* __restrict__ xsrc, const float* __restrict__ om,
           const bf16_t* __restrict__ afrag, const float* __restrict__ bias,
           float* __restrict__ out) {
    __shared__ float s_tap[8][144][8];
    int wid  = threadIdx.x >> 5;
    int lane = threadIdx.x & 31;
    int wave = blockIdx.x * 8 + wid;
    int b = wave >> 10, nt = wave & (NT - 1);
    int nbase = nt * 16;
    const float* omb = om + b * KK3 * HW;

    for (int r = lane; r < 144; r += 32) {        // phase A
        int t = r >> 4, n2 = r & 15;
        int p = nbase + n2, y = p >> 7, x = p & (HL - 1);
        float dyv = omb[t * HW + p];
        float dxv = omb[(9 + t) * HW + p];
        float mv  = omb[(18 + t) * HW + p];
        mv = 1.0f / (1.0f + __expf(-mv));         // sigmoid(mask)
        float py = (float)y + (float)(t / 3 - 1) + dyv;
        float px = (float)x + (float)(t % 3 - 1) + dxv;
        float y0f = floorf(py), x0f = floorf(px);
        float fy = py - y0f, fx = px - x0f;
        int iy0 = (int)y0f, ix0 = (int)x0f;
        float vy0 = (iy0 >= 0     && iy0     <= HL - 1) ? 1.f : 0.f;
        float vy1 = (iy0 + 1 >= 0 && iy0 + 1 <= HL - 1) ? 1.f : 0.f;
        float vx0 = (ix0 >= 0     && ix0     <= HL - 1) ? 1.f : 0.f;
        float vx1 = (ix0 + 1 >= 0 && ix0 + 1 <= HL - 1) ? 1.f : 0.f;
        int y0c = min(max(iy0, 0), HL - 1),  y1c = min(max(iy0 + 1, 0), HL - 1);
        int x0c = min(max(ix0, 0), HL - 1),  x1c = min(max(ix0 + 1, 0), HL - 1);
        s_tap[wid][r][0] = __int_as_float(y0c * HL + x0c);
        s_tap[wid][r][1] = __int_as_float(y0c * HL + x1c);
        s_tap[wid][r][2] = __int_as_float(y1c * HL + x0c);
        s_tap[wid][r][3] = __int_as_float(y1c * HL + x1c);
        s_tap[wid][r][4] = mv * (1.f - fy) * (1.f - fx) * vy0 * vx0;
        s_tap[wid][r][5] = mv * (1.f - fy) * fx         * vy0 * vx1;
        s_tap[wid][r][6] = mv * fy         * (1.f - fx) * vy1 * vx0;
        s_tap[wid][r][7] = mv * fy         * fx         * vy1 * vx1;
    }
    __syncthreads();

    int h = lane >> 4, n = lane & 15;
    const float* xb = xsrc + b * CC * HW;
    const v16bf* A = (const v16bf*)afrag;
    v8f acc[4] = {{}, {}, {}, {}};
    for (int kt = 0; kt < 18; ++kt) {             // phase B
        if (kt < 17)                               // warm next channel plane
            __builtin_prefetch(xb + (((kt + 1) * 32 + 16 * h) / 9) * HW + nbase, 0, 1);
        v8u bw;
        #pragma unroll
        for (int j = 0; j < 8; ++j) {
            float v[2];
            #pragma unroll
            for (int q = 0; q < 2; ++q) {
                int k = kt * 32 + 16 * h + 2 * j + q;
                int c = k / 9, t = k - 9 * c;
                const float* rec = s_tap[wid][t * 16 + n];
                const float* cp  = xb + c * HW;
                v[q] = rec[4] * cp[__float_as_int(rec[0])]
                     + rec[5] * cp[__float_as_int(rec[1])]
                     + rec[6] * cp[__float_as_int(rec[2])]
                     + rec[7] * cp[__float_as_int(rec[3])];
            }
            bw[j] = pack2bf(v[0], v[1]);
        }
        v16bf bfr = __builtin_bit_cast(v16bf, bw);
        #pragma unroll
        for (int mt = 0; mt < 4; ++mt)
            acc[mt] = wmma_bf16(A[(mt * 18 + kt) * 32 + lane], bfr, acc[mt]);
    }
    #pragma unroll
    for (int mt = 0; mt < 4; ++mt)
        #pragma unroll
        for (int r = 0; r < 8; ++r) {
            int m = mt * 16 + 8 * h + r;
            out[(b * CC + m) * HW + nbase + n] = acc[mt][r] + bias[m];
        }
}

// ---------------------------------------------------------------------------
extern "C" void kernel_launch(void* const* d_in, const int* in_sizes, int n_in,
                              void* d_out, int out_size, void* d_ws, size_t ws_size,
                              hipStream_t stream) {
    const float* high   = (const float*)d_in[0];
    const float* low    = (const float*)d_in[1];
    const float* w_off  = (const float*)d_in[2];
    const float* wh_om  = (const float*)d_in[3];
    const float* bh_om  = (const float*)d_in[4];
    const float* wh_dcn = (const float*)d_in[5];
    const float* bh_dcn = (const float*)d_in[6];
    const float* wl_om  = (const float*)d_in[7];
    const float* bl_om  = (const float*)d_in[8];
    const float* wl_dcn = (const float*)d_in[9];
    const float* bl_dcn = (const float*)d_in[10];

    // workspace layout (floats, then bf16 fragments)
    float* ws    = (float*)d_ws;
    float* hup   = ws;                         // 4*64*16384 = 4,194,304
    float* ofeat = ws + 4194304;               // 4,194,304
    float* om_h  = ws + 8388608;               // 4*27*16384 = 1,769,472
    float* om_l  = ws + 10158080;              // 1,769,472
    bf16_t* bfb   = (bf16_t*)(ws + 11927552);
    bf16_t* f_woff = bfb;                      //  4x4 tiles * 512 =  8192
    bf16_t* f_whom = bfb + 8192;               // 2x18 tiles * 512 = 18432
    bf16_t* f_wlom = bfb + 26624;              // 18432
    bf16_t* f_whd  = bfb + 45056;              // 4x18 tiles * 512 = 36864
    bf16_t* f_wld  = bfb + 81920;              // 36864

    // pack weights into WMMA A-fragment order (bf16)
    k_pack<<<(8192  + 255) / 256, 256, 0, stream>>>(w_off,  f_woff, 4,  4, 64, 128);
    k_pack<<<(18432 + 255) / 256, 256, 0, stream>>>(wh_om,  f_whom, 2, 18, 27, 576);
    k_pack<<<(18432 + 255) / 256, 256, 0, stream>>>(wl_om,  f_wlom, 2, 18, 27, 576);
    k_pack<<<(36864 + 255) / 256, 256, 0, stream>>>(wh_dcn, f_whd,  4, 18, 64, 576);
    k_pack<<<(36864 + 255) / 256, 256, 0, stream>>>(wl_dcn, f_wld,  4, 18, 64, 576);

    // pipeline
    k_upsample<<<(BB * CC * HW) / 256, 256, 0, stream>>>(high, hup);
    k_conv1x1 <<<BB * NT / 8, 256, 0, stream>>>(hup, low, f_woff, ofeat);
    k_om      <<<BB * NT / 8, 256, 0, stream>>>(ofeat, f_whom, bh_om, om_h);
    k_om      <<<BB * NT / 8, 256, 0, stream>>>(ofeat, f_wlom, bl_om, om_l);

    float* out = (float*)d_out;
    k_dcn<<<BB * NT / 8, 256, 0, stream>>>(hup, om_h, f_whd, bh_dcn, out);
    k_dcn<<<BB * NT / 8, 256, 0, stream>>>(low, om_l, f_wld, bl_dcn, out + 4194304);
}
